// GCNRegressor_15762529976888
// MI455X (gfx1250) — compile-verified, matrix operand order checked by reading, not run on hardware
//
#include <hip/hip_runtime.h>
#include <hip/hip_bf16.h>

// ---------------------------------------------------------------------------
// GCN regressor for MI455X (gfx1250).
//   h = relu( segment_sum( (h@W)[src] * norm ) + (h@W)*dinv^2 + b )  x3
//   pooled mean per graph -> 64-wide MLP -> scalar per graph
//
// Hidden state is stored as packed bf16 dwords (rounded at the producer --
// numerically identical to rounding at the WMMA input), so the GEMM inner
// loop is pure b128 loads + v_wmma_f32_16x16x32_bf16. C stores use one base
// pointer + constant immediate offsets (uniform tile-bound check hoisted).
// Sparse phase: HW fp32 global atomics into L2-resident accumulators.
// ---------------------------------------------------------------------------

typedef __attribute__((ext_vector_type(16))) __bf16   v16bf;
typedef __attribute__((ext_vector_type(8)))  float    v8f;
typedef __attribute__((ext_vector_type(8)))  unsigned v8u;

__device__ __forceinline__ unsigned pack2bf(float a, float b) {
#if __has_builtin(__builtin_amdgcn_cvt_pk_bf16_f32)
  typedef __attribute__((ext_vector_type(2))) __bf16 v2bf;
  v2bf p = __builtin_amdgcn_cvt_pk_bf16_f32(a, b);
  return __builtin_bit_cast(unsigned, p);
#else
  // round-to-nearest-even
  unsigned ua = __builtin_bit_cast(unsigned, a);
  unsigned ub = __builtin_bit_cast(unsigned, b);
  ua = (ua + 0x7fffu + ((ua >> 16) & 1u)) >> 16;
  ub = (ub + 0x7fffu + ((ub >> 16) & 1u)) & 0xffff0000u;
  return ua | ub;
#endif
}

// ---------------------------- utility kernels ------------------------------

__global__ void fill_kernel(float* __restrict__ p, float v, int n) {
  int i = blockIdx.x * blockDim.x + threadIdx.x;
  if (i < n) p[i] = v;
}

__global__ void edge_deg_kernel(const int* __restrict__ dst,
                                float* __restrict__ deg, int E) {
  int e = blockIdx.x * blockDim.x + threadIdx.x;
  if (e < E) unsafeAtomicAdd(&deg[dst[e]], 1.0f);
}

__global__ void rsqrt_kernel(const float* __restrict__ deg,
                             float* __restrict__ dinv, int n) {
  int i = blockIdx.x * blockDim.x + threadIdx.x;
  if (i < n) dinv[i] = rsqrtf(deg[i]);
}

// f32 rows -> packed bf16 pairs (row-major; fragment-order for the GEMM)
__global__ void pack_rows_kernel(const float* __restrict__ in,
                                 unsigned* __restrict__ outbf, int total_dw) {
  int i = blockIdx.x * blockDim.x + threadIdx.x;
  if (i >= total_dw) return;
  outbf[i] = pack2bf(in[2 * i], in[2 * i + 1]);
}

// --------------------- W pre-pack into fragment layout ---------------------
// wp[((kc*8 + nt)*32 + lane)*8 + v] = bf16 pair (k, k+1) for
//   k = kc*32 + (v>>2)*16 + khalf*8 + (v&3)*2,  n = nt*16 + (lane&15)
__global__ __launch_bounds__(256)
void wpack_kernel(const float* __restrict__ W, unsigned* __restrict__ wp) {
  int i = blockIdx.x * blockDim.x + threadIdx.x;
  if (i >= 8192) return;
  int v    = i & 7;
  int lane = (i >> 3) & 31;
  int nt   = (i >> 8) & 7;
  int kc   = (i >> 11) & 3;
  int khalf = (lane >> 4) & 1;
  int k = kc * 32 + (v >> 2) * 16 + khalf * 8 + (v & 3) * 2;
  int n = nt * 16 + (lane & 15);
  wp[i] = pack2bf(W[k * 128 + n], W[(k + 1) * 128 + n]);
}

// ------------------------------ dense GEMM ---------------------------------
// C[M,128] = A[M,128] x W[128,128], A given as packed bf16 dwords (64/row).
// One wave per 16-row block; A fragments loaded once (pure b128 loads, no
// conversion) and reused across all 8 column tiles -> 32 v_wmma per wave.
__global__ __launch_bounds__(256)
void gemm_wmma_kernel(const unsigned* __restrict__ Abf,
                      const unsigned* __restrict__ wp,
                      float* __restrict__ C, int M) {
  int wave = (int)((blockIdx.x * (unsigned)blockDim.x + threadIdx.x) >> 5);
  int lane = threadIdx.x & 31;
  int m0 = wave * 16;
  if (m0 >= M) return;

  int arow = m0 + (lane & 15);
  if (arow >= M) arow = M - 1;           // safety clamp (M % 16 == 0 normally)
  int khalf = (lane >> 4) & 1;
  const unsigned* arowp = Abf + (size_t)arow * 64;

  // fragment dword v <-> row dword index kc*16 + (v>>2)*8 + khalf*4 + (v&3)
  v16bf afrag[4];
#pragma unroll
  for (int kc = 0; kc < 4; ++kc) {
    const uint4* ap = (const uint4*)(arowp + kc * 16 + khalf * 4);
    uint4 lo = ap[0];   // v = 0..3
    uint4 hi = ap[2];   // v = 4..7  (+8 dwords)
    v8u u;
    u[0] = lo.x; u[1] = lo.y; u[2] = lo.z; u[3] = lo.w;
    u[4] = hi.x; u[5] = hi.y; u[6] = hi.z; u[7] = hi.w;
    afrag[kc] = __builtin_bit_cast(v16bf, u);
  }

  int nlane = lane & 15;
  // single base pointer; all store offsets below are compile-time constants
  float* cbase = C + (size_t)(m0 + khalf * 8) * 128 + nlane;
  bool full = (m0 + 16) <= M;            // wave-uniform

#pragma unroll
  for (int nt = 0; nt < 8; ++nt) {
    v8f acc = {0.f, 0.f, 0.f, 0.f, 0.f, 0.f, 0.f, 0.f};
#pragma unroll
    for (int kc = 0; kc < 4; ++kc) {
      const uint4* bp = (const uint4*)(wp + ((size_t)((kc * 8 + nt) * 32 + lane)) * 8);
      uint4 b0 = bp[0];
      uint4 b1 = bp[1];
      v8u u;
      u[0] = b0.x; u[1] = b0.y; u[2] = b0.z; u[3] = b0.w;
      u[4] = b1.x; u[5] = b1.y; u[6] = b1.z; u[7] = b1.w;
      v16bf bfrag = __builtin_bit_cast(v16bf, u);
      acc = __builtin_amdgcn_wmma_f32_16x16x32_bf16(
          /*neg_a=*/false, afrag[kc], /*neg_b=*/false, bfrag,
          /*c_mod=*/(short)0, acc, /*reuse_a=*/false, /*reuse_b=*/false);
    }
    // C/D layout: lane holds col n; VGPR v holds row v + 8*khalf
    if (full) {
#pragma unroll
      for (int v = 0; v < 8; ++v)
        cbase[nt * 16 + v * 128] = acc[v];          // immediate offsets
    } else {
#pragma unroll
      for (int v = 0; v < 8; ++v) {
        int r = m0 + v + 8 * khalf;
        if (r < M) cbase[nt * 16 + v * 128] = acc[v];
      }
    }
  }
}

// --------------------------- sparse edge phase -----------------------------
// One wave per edge; lane handles a float4 (512B coalesced row), 4 HW fp32
// atomics into agg[dst] which is L2-resident (51 MB < 192 MB L2).
__global__ __launch_bounds__(256)
void edge_scatter_kernel(const float* __restrict__ t, const int* __restrict__ src,
                         const int* __restrict__ dst, const float* __restrict__ dinv,
                         float* __restrict__ agg, int E) {
  size_t gtid = blockIdx.x * (size_t)blockDim.x + threadIdx.x;
  int e = (int)(gtid >> 5);
  if (e >= E) return;
  int lane = threadIdx.x & 31;
  int s = src[e];
  int d = dst[e];
  float nrm = dinv[s] * dinv[d];
  float4 v = ((const float4*)(t + (size_t)s * 128))[lane];
  float* ad = agg + (size_t)d * 128 + lane * 4;
  unsafeAtomicAdd(ad + 0, v.x * nrm);
  unsafeAtomicAdd(ad + 1, v.y * nrm);
  unsafeAtomicAdd(ad + 2, v.z * nrm);
  unsafeAtomicAdd(ad + 3, v.w * nrm);
}

// h_out(bf16 packed) = relu(agg + t * dinv^2 + b)   [layers 0,1]
__global__ void relu_bias_bf16_kernel(const float* __restrict__ agg,
                                      const float* __restrict__ t,
                                      const float* __restrict__ dinv,
                                      const float* __restrict__ b,
                                      unsigned* __restrict__ outbf, int total_dw) {
  int i = blockIdx.x * blockDim.x + threadIdx.x;
  if (i >= total_dw) return;
  int node = i >> 6;
  int c = (i & 63) * 2;
  size_t idx = (size_t)node * 128 + c;
  float di = dinv[node];
  float d2 = di * di;
  float v0 = agg[idx]     + t[idx]     * d2 + b[c];
  float v1 = agg[idx + 1] + t[idx + 1] * d2 + b[c + 1];
  outbf[i] = pack2bf(fmaxf(v0, 0.f), fmaxf(v1, 0.f));
}

// h_out(f32, in-place on agg) = relu(agg + t * dinv^2 + b)   [layer 2]
__global__ void relu_bias_f32_kernel(float* __restrict__ agg,
                                     const float* __restrict__ t,
                                     const float* __restrict__ dinv,
                                     const float* __restrict__ b, int total) {
  int i = blockIdx.x * blockDim.x + threadIdx.x;
  if (i >= total) return;
  int node = i >> 7;
  int c = i & 127;
  float di = dinv[node];
  float v = agg[i] + t[i] * di * di + b[c];
  agg[i] = v > 0.f ? v : 0.f;
}

// ------------------------------- pooling -----------------------------------
__global__ __launch_bounds__(256)
void pool_scatter_kernel(const float* __restrict__ h, const int* __restrict__ batch,
                         float* __restrict__ pooled, float* __restrict__ counts,
                         int N) {
  size_t gtid = blockIdx.x * (size_t)blockDim.x + threadIdx.x;
  int node = (int)(gtid >> 5);
  if (node >= N) return;
  int lane = threadIdx.x & 31;
  int g = batch[node];
  float4 v = ((const float4*)(h + (size_t)node * 128))[lane];
  float* p = pooled + (size_t)g * 128 + lane * 4;
  unsafeAtomicAdd(p + 0, v.x);
  unsafeAtomicAdd(p + 1, v.y);
  unsafeAtomicAdd(p + 2, v.z);
  unsafeAtomicAdd(p + 3, v.w);
  if (lane == 0) unsafeAtomicAdd(&counts[g], 1.0f);
}

// ------------------------------- MLP head ----------------------------------
__global__ __launch_bounds__(64)
void mlp_head_kernel(const float* __restrict__ pooled, const float* __restrict__ counts,
                     const float* __restrict__ lw1, const float* __restrict__ lb1,
                     const float* __restrict__ lw2, const float* __restrict__ lb2,
                     float* __restrict__ out) {
  int g = blockIdx.x;
  int j = threadIdx.x;
  __shared__ float hid[64];
  float inv = 1.0f / fmaxf(counts[g], 1.0f);
  float acc = lb1[j];
#pragma unroll 4
  for (int k = 0; k < 128; ++k)
    acc = fmaf(pooled[(size_t)g * 128 + k] * inv, lw1[k * 64 + j], acc);
  hid[j] = fmaxf(acc, 0.f);
  __syncthreads();
  if (j == 0) {
    float s = lb2[0];
#pragma unroll 8
    for (int k = 0; k < 64; ++k) s = fmaf(hid[k], lw2[k], s);
    out[g] = s;
  }
}

// ------------------------------- launcher ----------------------------------
extern "C" void kernel_launch(void* const* d_in, const int* in_sizes, int n_in,
                              void* d_out, int out_size, void* d_ws, size_t ws_size,
                              hipStream_t stream) {
  const float* x     = (const float*)d_in[0];
  const int*   ei    = (const int*)d_in[1];
  const int*   batch = (const int*)d_in[2];
  const float* Wl[3] = {(const float*)d_in[3], (const float*)d_in[5], (const float*)d_in[7]};
  const float* bl[3] = {(const float*)d_in[4], (const float*)d_in[6], (const float*)d_in[8]};
  const float* lw1 = (const float*)d_in[9];
  const float* lb1 = (const float*)d_in[10];
  const float* lw2 = (const float*)d_in[11];
  const float* lb2 = (const float*)d_in[12];
  float* out = (float*)d_out;

  const int N = in_sizes[0] / 128;   // 100000
  const int E = in_sizes[1] / 2;     // 3200000
  const int G = out_size;            // 4096
  const size_t NB = (size_t)N * 128;
  const int NDW = (int)(NB / 2);     // packed bf16 dwords per feature matrix

  float* ws = (float*)d_ws;
  float*    t      = ws;                                   // NB f32
  float*    agg    = ws + NB;                              // NB f32
  float*    deg    = ws + 2 * NB;                          // N
  float*    dinv   = deg + N;                              // N
  float*    pooled = dinv + N;                             // G*128
  float*    counts = pooled + (size_t)G * 128;             // G
  unsigned* wpack  = (unsigned*)(counts + G);              // 8192 dw (32 KB)
  unsigned* bfA    = wpack + 8192;                         // NB/2 dw
  unsigned* bfB    = bfA + NDW;                            // NB/2 dw

  const int* src = ei;
  const int* dst = ei + E;

  const int TB = 256;
  // --- degree / normalization ---
  fill_kernel<<<(N + TB - 1) / TB, TB, 0, stream>>>(deg, 1.0f, N);   // self-loops
  edge_deg_kernel<<<(E + TB - 1) / TB, TB, 0, stream>>>(dst, deg, E);
  rsqrt_kernel<<<(N + TB - 1) / TB, TB, 0, stream>>>(deg, dinv, N);

  const int mtiles = (N + 15) / 16;
  const unsigned gemm_blocks = (unsigned)(((size_t)mtiles * 32 + TB - 1) / TB);
  const unsigned edge_blocks = (unsigned)(((size_t)E * 32 + TB - 1) / TB);
  const unsigned feat_blocks = (unsigned)((NB + TB - 1) / TB);
  const unsigned dw_blocks   = (unsigned)((NDW + TB - 1) / TB);

  // pack input features once: x -> bfA
  pack_rows_kernel<<<dw_blocks, TB, 0, stream>>>(x, bfA, NDW);

  // layers 0,1: bf16 in -> bf16 out (ping-pong bfA/bfB); layer 2: f32 out
  unsigned* bfin  = bfA;
  unsigned* bfout = bfB;
  for (int l = 0; l < 3; ++l) {
    wpack_kernel<<<32, TB, 0, stream>>>(Wl[l], wpack);
    gemm_wmma_kernel<<<gemm_blocks, TB, 0, stream>>>(bfin, wpack, t, N);
    fill_kernel<<<feat_blocks, TB, 0, stream>>>(agg, 0.0f, (int)NB);
    edge_scatter_kernel<<<edge_blocks, TB, 0, stream>>>(t, src, dst, dinv, agg, E);
    if (l < 2) {
      relu_bias_bf16_kernel<<<dw_blocks, TB, 0, stream>>>(agg, t, dinv, bl[l], bfout, NDW);
      unsigned* tmp = bfin; bfin = bfout; bfout = tmp;
    } else {
      relu_bias_f32_kernel<<<feat_blocks, TB, 0, stream>>>(agg, t, dinv, bl[l], (int)NB);
    }
  }

  // --- global mean pool + MLP head (h3 lives in agg as f32) ---
  fill_kernel<<<(G * 128 + TB - 1) / TB, TB, 0, stream>>>(pooled, 0.0f, G * 128);
  fill_kernel<<<(G + TB - 1) / TB, TB, 0, stream>>>(counts, 0.0f, G);
  const unsigned pool_blocks = (unsigned)(((size_t)N * 32 + TB - 1) / TB);
  pool_scatter_kernel<<<pool_blocks, TB, 0, stream>>>(agg, batch, pooled, counts, N);
  mlp_head_kernel<<<G, 64, 0, stream>>>(pooled, counts, lw1, lb1, lw2, lb2, out);
}
